// NetContPID_OhSoft_28157805592652
// MI455X (gfx1250) — matrix operand order repaired and checked by reading, not compiled, not run on hardware
//
#include <hip/hip_runtime.h>

typedef __attribute__((ext_vector_type(16))) _Float16 v16h;
typedef __attribute__((ext_vector_type(8)))  float    v8f;

#define NBOUNDS 63
#define NDISC   64
#define NHID    (NDISC * NDISC * NDISC)   // 262144

// One wave (32 lanes) handles a tile of 16 samples.
// mu[b,o] = sum_k w3[k] * W_mu[o, flat(d(b)+off(k))]  (27 taps, boundary-clipped)
// Recast as GEMM: G[b, 2k+o] (K=54, pad 64)  @  B[j,o] = w3[j/2]*(j&1==o).
// G split into f16 hi + f16 residual -> 4x v_wmma_f32_16x16x32_f16, f32 accurate.
__global__ __launch_bounds__(32) void ohsoft_wmma_kernel(
    const float* __restrict__ x,       // [nb, 3]
    const float* __restrict__ bounds,  // [63]
    const float* __restrict__ kern,    // [27] (1,1,3,3,3 flattened)
    const float* __restrict__ Wmu,     // [2, 64^3]
    float* __restrict__ out,           // [nb, 2]
    int nb)
{
    __shared__ float sb[NBOUNDS];
    __shared__ float sk[27];

    const int lane = threadIdx.x;          // 0..31
    for (int i = lane; i < NBOUNDS; i += 32) sb[i] = bounds[i];
    if (lane < 27) sk[lane] = kern[lane];
    __syncthreads();

    const int  m       = lane & 15;                 // sample within tile (dup on upper half)
    int        s       = blockIdx.x * 16 + m;       // global sample
    const bool hi_half = lane >= 16;
    const int  N       = lane & 15;                 // WMMA column for B/D
    if (s >= nb) s = nb - 1;                        // clamp (WMMA needs full EXEC)

    // --- bucketize: searchsorted side='left' == count(bounds < x) ---
    int d0 = 0, d1 = 0, d2 = 0;
    {
        const float x0 = x[s * 3 + 0];
        const float x1 = x[s * 3 + 1];
        const float x2 = x[s * 3 + 2];
        #pragma unroll
        for (int j = 0; j < NBOUNDS; ++j) {
            const float bv = sb[j];
            d0 += (bv < x0) ? 1 : 0;
            d1 += (bv < x1) ? 1 : 0;
            d2 += (bv < x2) ? 1 : 0;
        }
    }

    // --- build A (hi/lo compensated) and constant B for two K-chunks of 32 ---
    v16h a_hi0, a_lo0, b0, a_hi1, a_lo1, b1;
    #pragma unroll
    for (int chunk = 0; chunk < 2; ++chunk) {
        const int cbase = chunk * 32;
        #pragma unroll
        for (int t = 0; t < 16; ++t) {
            // A layout (16-bit 16x32): lanes<16 hold K {0..7,16..23}, lanes>=16 {8..15,24..31}
            const int KA = (t < 8 ? t : 8 + t) + (hi_half ? 8 : 0);
            const int jA = cbase + KA;               // 0..63 (valid < 54)
            float g = 0.0f;
            if (jA < 54) {
                const int k  = jA >> 1;              // tap 0..26
                const int o  = jA & 1;               // output 0..1
                const int di = k / 9, rem = k % 9, dj = rem / 3, dk = rem % 3;
                const int c0 = d0 + di - 1, c1 = d1 + dj - 1, c2 = d2 + dk - 1;
                if ((unsigned)c0 < 64u && (unsigned)c1 < 64u && (unsigned)c2 < 64u) {
                    const int flat = (c0 * 64 + c1) * 64 + c2;
                    g = Wmu[o * NHID + flat];
                }
            }
            const _Float16 gh = (_Float16)g;
            const _Float16 gl = (_Float16)(g - (float)gh);

            // B layout (16-bit 32x16): lanes<16 hold K {0..7,16..23}, lanes>=16 {8..15,24..31}, N=lane%16
            const int jB = cbase + (t < 8 ? t : 8 + t) + (hi_half ? 8 : 0);
            _Float16 bv = (_Float16)0.0f;
            if (jB < 54 && (jB & 1) == N)
                bv = (_Float16)sk[jB >> 1];          // products of {0.25,0.5}: exact in f16

            if (chunk == 0) { a_hi0[t] = gh; a_lo0[t] = gl; b0[t] = bv; }
            else            { a_hi1[t] = gh; a_lo1[t] = gl; b1[t] = bv; }
        }
    }

    // --- contraction: 4x WMMA, f32 accumulate ---
    v8f acc = {};
    acc = __builtin_amdgcn_wmma_f32_16x16x32_f16(false, a_hi0, false, b0, (short)0, acc, false, false);
    acc = __builtin_amdgcn_wmma_f32_16x16x32_f16(false, a_lo0, false, b0, (short)0, acc, false, false);
    acc = __builtin_amdgcn_wmma_f32_16x16x32_f16(false, a_hi1, false, b1, (short)0, acc, false, false);
    acc = __builtin_amdgcn_wmma_f32_16x16x32_f16(false, a_lo1, false, b1, (short)0, acc, false, false);

    // --- store: D f32 16x16 layout: lane -> N (and +16 -> M+8), VGPR v -> M ---
    if (N < 2) {
        #pragma unroll
        for (int v = 0; v < 8; ++v) {
            const int M   = v + (hi_half ? 8 : 0);
            const int row = blockIdx.x * 16 + M;
            if (row < nb) out[row * 2 + N] = acc[v];
        }
    }
}

extern "C" void kernel_launch(void* const* d_in, const int* in_sizes, int n_in,
                              void* d_out, int out_size, void* d_ws, size_t ws_size,
                              hipStream_t stream) {
    (void)n_in; (void)out_size; (void)d_ws; (void)ws_size;
    const float* x      = (const float*)d_in[0];   // [256,3]
    const float* bounds = (const float*)d_in[1];   // [63]
    const float* kern   = (const float*)d_in[2];   // [27]
    const float* Wmu    = (const float*)d_in[3];   // [2, 262144]
    float*       out    = (float*)d_out;           // [256,2]

    const int nb     = in_sizes[0] / 3;            // 256
    const int blocks = (nb + 15) / 16;             // 16 waves total
    ohsoft_wmma_kernel<<<blocks, 32, 0, stream>>>(x, bounds, kern, Wmu, out, nb);
}
